// Flashback_46308337385592
// MI455X (gfx1250) — matrix operand order repaired and checked by reading, not compiled, not run on hardware
//
#include <hip/hip_runtime.h>
#include <hip/hip_bf16.h>
#include <math.h>

#define S_LEN 20
#define B_LEN 256
#define H_DIM 128
#define L_LOC 20000
#define K_DIM 256                   // 2*H
#define M_ROWS (S_LEN * B_LEN)      // 5120
#define MT_CNT (M_ROWS / 16)        // 320
#define NT_PAD 1256                 // 157 * 8 (covers 20000 cols, padded)
#define KSTEPS (K_DIM / 32)         // 8
#define DAY 86400.0f

typedef __attribute__((ext_vector_type(16))) __bf16 v16bf;
typedef __attribute__((ext_vector_type(8)))  float  v8f;

static __device__ __forceinline__ unsigned short f32_to_bf16(float f) {
  unsigned int u = __float_as_uint(f);
  u += 0x7fffu + ((u >> 16) & 1u);      // round-to-nearest-even
  return (unsigned short)(u >> 16);
}

// ---------- x_emb: gathered rows of D^-1(A+I) @ enc_emb (COO structure known) ----------
__global__ void k_xemb(const int* __restrict__ x, const float* __restrict__ enc,
                       const int* __restrict__ gcol, const float* __restrict__ gval,
                       float* __restrict__ xemb) {
  int p = blockIdx.x;               // s*B + b
  int h = threadIdx.x;              // 0..127
  __shared__ int   cols[11];
  __shared__ float vals[11];
  int r = x[p];
  if (h < 10)       { cols[h]  = gcol[r * 10 + h];       vals[h]  = gval[r * 10 + h]; }
  else if (h == 10) { cols[10] = gcol[L_LOC * 10 + r];   vals[10] = gval[L_LOC * 10 + r]; }
  __syncthreads();
  float acc = 0.f;
#pragma unroll
  for (int k = 0; k < 11; ++k) acc += vals[k] * enc[(size_t)cols[k] * H_DIM + h];
  xemb[(size_t)p * H_DIM + h] = acc;
}

// ---------- user_pref for the 256 active users only (20 nnz/row) ----------
__global__ void k_upref(const int* __restrict__ au, const float* __restrict__ enc,
                        const int* __restrict__ icol, const float* __restrict__ ival,
                        float* __restrict__ upref) {
  int b = blockIdx.x;
  int h = threadIdx.x;
  __shared__ int   cols[20];
  __shared__ float vals[20];
  int u = au[b];
  if (h < 20) { cols[h] = icol[u * 20 + h]; vals[h] = ival[u * 20 + h]; }
  __syncthreads();
  float acc = 0.f;
#pragma unroll
  for (int k = 0; k < 20; ++k) acc += vals[k] * enc[(size_t)cols[k] * H_DIM + h];
  upref[b * H_DIM + h] = acc;
}

// ---------- user_loc_sim = exp(-||upref - xemb||) ----------
__global__ void k_sim(const float* __restrict__ upref, const float* __restrict__ xemb,
                      float* __restrict__ sim) {
  int p = blockIdx.x;               // s*B + b
  int tid = threadIdx.x;
  int b = p & (B_LEN - 1);
  __shared__ float red[H_DIM];
  float d = upref[b * H_DIM + tid] - xemb[(size_t)p * H_DIM + tid];
  red[tid] = d * d;
  __syncthreads();
  for (int st = 64; st > 0; st >>= 1) {
    if (tid < st) red[tid] += red[tid + st];
    __syncthreads();
  }
  if (tid == 0) sim[p] = expf(-sqrtf(red[0]));
}

// ---------- transpose RNN weights into ws (coalesced reads in RNN) ----------
__global__ void k_wt(const float* __restrict__ Wih, const float* __restrict__ Whh,
                     float* __restrict__ wtih, float* __restrict__ wthh) {
  int idx = blockIdx.x * blockDim.x + threadIdx.x;   // 0..16383
  int j = idx >> 7, k = idx & 127;
  wtih[k * H_DIM + j] = Wih[idx];
  wthh[k * H_DIM + j] = Whh[idx];
}

// ---------- RNN: one block per batch element (independent across b) ----------
__global__ void k_rnn(const float* __restrict__ xemb, const float* __restrict__ h0,
                      const float* __restrict__ wtih, const float* __restrict__ wthh,
                      const float* __restrict__ bih, const float* __restrict__ bhh,
                      float* __restrict__ rnnout, float* __restrict__ hlast) {
  int b = blockIdx.x;
  int j = threadIdx.x;
  __shared__ float xs[H_DIM], hc[H_DIM];
  hc[j] = h0[b * H_DIM + j];
  float bias = bih[j] + bhh[j];
  for (int t = 0; t < S_LEN; ++t) {
    xs[j] = xemb[((size_t)t * B_LEN + b) * H_DIM + j];
    __syncthreads();
    float acc = bias;
#pragma unroll 4
    for (int k = 0; k < H_DIM; ++k)
      acc += xs[k] * wtih[k * H_DIM + j] + hc[k] * wthh[k * H_DIM + j];
    float v = tanhf(acc);
    rnnout[((size_t)t * B_LEN + b) * H_DIM + j] = v;
    __syncthreads();
    hc[j] = v;
    __syncthreads();
  }
  hlast[b * H_DIM + j] = hc[j];
}

// ---------- causal decay-weighted average of RNN outputs ----------
__global__ void k_outw(const float* __restrict__ t, const float* __restrict__ s,
                       const float* __restrict__ sim, const float* __restrict__ rnnout,
                       float* __restrict__ outw) {
  int p = blockIdx.x;               // i*B + b
  int tid = threadIdx.x;
  int i = p / B_LEN, b = p % B_LEN;
  __shared__ float wj[32];
  __shared__ float sumw;
  if (tid <= i) {
    int j = tid;
    float dt = t[i * B_LEN + b] - t[j * B_LEN + b];
    float a  = (cosf(dt * (2.f * 3.14159265358979323846f / DAY)) + 1.f) * 0.5f
             * expf(-dt / DAY * 0.1f);
    float dx = s[(i * B_LEN + b) * 2 + 0] - s[(j * B_LEN + b) * 2 + 0];
    float dy = s[(i * B_LEN + b) * 2 + 1] - s[(j * B_LEN + b) * 2 + 1];
    float bb = expf(-sqrtf(dx * dx + dy * dy) * 100.f);
    wj[j] = (a * bb + 1e-10f) * sim[j * B_LEN + b];
  }
  __syncthreads();
  if (tid == 0) {
    float sw = 0.f;
    for (int j = 0; j <= i; ++j) sw += wj[j];
    sumw = sw;
  }
  __syncthreads();
  float acc = 0.f;
  for (int j = 0; j <= i; ++j)
    acc += wj[j] * rnnout[((size_t)j * B_LEN + b) * H_DIM + tid];
  outw[(size_t)p * H_DIM + tid] = acc / sumw;
}

// ---------- pack A = [out_w | user_emb[au]] into WMMA 16-bit A lane layout ----------
// 16-bit A 16x32: lane l (half = l/16, m = l%16); VGPR v<4: K = half*8 + 2v + {0,1};
// VGPR v>=4: K = 16 + half*8 + 2(v-4) + {0,1}.
__global__ void k_packA(const float* __restrict__ outw, const float* __restrict__ uemb,
                        const int* __restrict__ au, unsigned short* __restrict__ Apk) {
  int mt = blockIdx.x / KSTEPS, ks = blockIdx.x % KSTEPS;
  int lane = threadIdx.x;
  int half = lane >> 4;
  int m = mt * 16 + (lane & 15);
  int b = m & (B_LEN - 1);
  int urow = au[b];
  size_t base = ((size_t)(mt * KSTEPS + ks) * 32 + lane) * 16;
#pragma unroll
  for (int j = 0; j < 16; ++j) {
    int pr = j >> 1, hi = j & 1;
    int k = (pr < 4) ? (ks * 32 + half * 8 + 2 * pr + hi)
                     : (ks * 32 + 16 + half * 8 + 2 * (pr - 4) + hi);
    float f = (k < H_DIM) ? outw[(size_t)m * H_DIM + k]
                          : uemb[(size_t)urow * H_DIM + (k - H_DIM)];
    Apk[base + j] = f32_to_bf16(f);
  }
}

// ---------- pack B = fc_W^T tiles into WMMA 16-bit B lane layout ----------
// B 32x16: lane l -> N = l%16; element j<8: K = (l/16)*8 + j; j>=8: K = 16 + (l/16)*8 + (j-8). Pad N >= 20000 with 0.
__global__ void k_packB(const float* __restrict__ fcW, unsigned short* __restrict__ Bpk) {
  int nt = blockIdx.x / KSTEPS, ks = blockIdx.x % KSTEPS;
  int lane = threadIdx.x;
  int n = nt * 16 + (lane & 15);
  size_t base = ((size_t)(nt * KSTEPS + ks) * 32 + lane) * 16;
#pragma unroll
  for (int j = 0; j < 16; ++j) {
    int k = ks * 32 + ((j < 8) ? ((lane >> 4) * 8 + j) : (16 + (lane >> 4) * 8 + (j - 8)));
    unsigned short v = 0;
    if (n < L_LOC) v = f32_to_bf16(fcW[(size_t)n * K_DIM + k]);
    Bpk[base + j] = v;
  }
}

// ---------- the big GEMM: y = A @ fc_W^T + fc_b via v_wmma_f32_16x16x32_bf16 ----------
__global__ __launch_bounds__(256) void k_gemm(const unsigned short* __restrict__ Apk,
                                              const unsigned short* __restrict__ Bpk,
                                              const float* __restrict__ fcb,
                                              float* __restrict__ y) {
  int lane = threadIdx.x & 31;
  int w = threadIdx.x >> 5;               // 0..7
  int wm = w >> 1, wn = w & 1;
  int mt = blockIdx.x * 4 + wm;           // < 320
  int ntbase = blockIdx.y * 8 + wn * 4;   // < 1256
  v8f acc[4] = {};
#pragma unroll
  for (int ks = 0; ks < KSTEPS; ++ks) {
    v16bf a = *(const v16bf*)(Apk + ((size_t)(mt * KSTEPS + ks) * 32 + lane) * 16);
    if (ks + 1 < KSTEPS)
      __builtin_prefetch(Apk + ((size_t)(mt * KSTEPS + ks + 1) * 32 + lane) * 16, 0, 3);
#pragma unroll
    for (int tnt = 0; tnt < 4; ++tnt) {
      v16bf bt = *(const v16bf*)(Bpk +
                  ((size_t)((ntbase + tnt) * KSTEPS + ks) * 32 + lane) * 16);
      acc[tnt] = __builtin_amdgcn_wmma_f32_16x16x32_bf16(
          false, a, false, bt, (short)0, acc[tnt], false, false);
    }
  }
  // C/D layout: VGPR r, lane l -> M = r + 8*(l/16), N = l%16
  int half = lane >> 4, nn = lane & 15;
#pragma unroll
  for (int tnt = 0; tnt < 4; ++tnt) {
    int n = (ntbase + tnt) * 16 + nn;
    if (n < L_LOC) {
      float bias = fcb[n];
#pragma unroll
      for (int r = 0; r < 8; ++r) {
        int m = mt * 16 + 8 * half + r;
        y[(size_t)m * L_LOC + n] = acc[tnt][r] + bias;
      }
    }
  }
}

extern "C" void kernel_launch(void* const* d_in, const int* in_sizes, int n_in,
                              void* d_out, int out_size, void* d_ws, size_t ws_size,
                              hipStream_t stream) {
  (void)in_sizes; (void)n_in; (void)out_size; (void)ws_size;
  const int*   x     = (const int*)  d_in[0];
  const float* t     = (const float*)d_in[1];
  const float* s     = (const float*)d_in[2];
  const int*   au    = (const int*)  d_in[3];
  const float* h0    = (const float*)d_in[4];
  const float* enc   = (const float*)d_in[5];
  const float* uemb  = (const float*)d_in[6];
  const float* Wih   = (const float*)d_in[7];
  const float* Whh   = (const float*)d_in[8];
  const float* bih   = (const float*)d_in[9];
  const float* bhh   = (const float*)d_in[10];
  const float* fcW   = (const float*)d_in[11];
  const float* fcb   = (const float*)d_in[12];
  const int*   gcol  = (const int*)  d_in[14];
  const float* gval  = (const float*)d_in[15];
  const int*   icol  = (const int*)  d_in[17];
  const float* ival  = (const float*)d_in[18];

  float* y = (float*)d_out;                               // (S,B,L)
  float* hlast = y + (size_t)M_ROWS * L_LOC;              // (1,B,H)

  char* ws = (char*)d_ws;
  float* xemb   = (float*)(ws + 0);                       // 2,621,440 B
  float* upref  = (float*)(ws + 2621440);                 //   131,072 B
  float* sim    = (float*)(ws + 2752512);                 //    20,480 B
  float* rnnout = (float*)(ws + 2772992);                 // 2,621,440 B
  float* outw   = (float*)(ws + 5394432);                 // 2,621,440 B
  float* wtih   = (float*)(ws + 8015872);                 //    65,536 B
  float* wthh   = (float*)(ws + 8081408);                 //    65,536 B
  unsigned short* Apk = (unsigned short*)(ws + 8146944);  // 2,621,440 B
  unsigned short* Bpk = (unsigned short*)(ws + 10768384); // 10,289,152 B

  k_xemb <<<M_ROWS, H_DIM, 0, stream>>>(x, enc, gcol, gval, xemb);
  k_upref<<<B_LEN,  H_DIM, 0, stream>>>(au, enc, icol, ival, upref);
  k_sim  <<<M_ROWS, H_DIM, 0, stream>>>(upref, xemb, sim);
  k_wt   <<<128,    H_DIM, 0, stream>>>(Wih, Whh, wtih, wthh);
  k_rnn  <<<B_LEN,  H_DIM, 0, stream>>>(xemb, h0, wtih, wthh, bih, bhh, rnnout, hlast);
  k_outw <<<M_ROWS, H_DIM, 0, stream>>>(t, s, sim, rnnout, outw);
  k_packA<<<MT_CNT * KSTEPS, 32, 0, stream>>>(outw, uemb, au, Apk);
  k_packB<<<NT_PAD * KSTEPS, 32, 0, stream>>>(fcW, Bpk);
  k_gemm <<<dim3(MT_CNT / 4, NT_PAD / 8), 256, 0, stream>>>(Apk, Bpk, fcb, y);
}
